// GCN_POSE_64527588655345
// MI455X (gfx1250) — compile-verified, matrix operand order checked by reading, not run on hardware
//
#include <hip/hip_runtime.h>

typedef __attribute__((ext_vector_type(16))) _Float16 v16h;
typedef __attribute__((ext_vector_type(8)))  float    v8f;

static constexpr float BN_SCALE_C = 0.99999500003749965f; // 1/sqrt(1+1e-5)

// ---------------------------------------------------------------------------
// WMMA fragment load (A 16x32 f16 per-lane layout, ISA 7.12.2):
// lane<16 holds K {0..7,16..23}, lane>=16 holds {8..15,24..31} of its row.
// Compiler coalesces the 8 dword loads into 2x global_load_b128.
// ---------------------------------------------------------------------------
__device__ inline void frag_ld(const _Float16* __restrict__ row, int k0, int g, v16h& f)
{
#pragma unroll
    for (int j = 0; j < 8; ++j) {
        const int kk = k0 + ((j < 4) ? (2 * j) : (2 * j + 8)) + 8 * g;
        union { unsigned u; _Float16 h[2]; } c;
        c.u = *(const unsigned*)(row + kk);
        f[2 * j] = c.h[0]; f[2 * j + 1] = c.h[1];
    }
}

// One 16x16 tile, K compile-time => fully unrolled, software-pipelineable.
template<int KP>
__device__ inline v8f tile_mm(const _Float16* __restrict__ Abase,
                              const _Float16* __restrict__ Bbase, int lane)
{
    const int g  = (lane >> 4) & 1;
    const int mr = lane & 15;
    const _Float16* ar = Abase + mr * KP;
    const _Float16* br = Bbase + mr * KP;
    v8f acc;
#pragma unroll
    for (int q = 0; q < 8; ++q) acc[q] = 0.f;
#pragma unroll
    for (int k0 = 0; k0 < KP; k0 += 32) {
        if (k0 + 96 < KP) {                       // constant-folds after unroll
            __builtin_prefetch(ar + k0 + 96, 0, 1);
            __builtin_prefetch(br + k0 + 96, 0, 1);
        }
        v16h a, b;
        frag_ld(ar, k0, g, a);
        frag_ld(br, k0, g, b);
        acc = __builtin_amdgcn_wmma_f32_16x16x32_f16(false, a, false, b,
                                                     (short)0, acc, false, false);
    }
    return acc;
}

// Three tiles sharing one A tile (GRU r/z/n gates): A loaded once per K-step,
// three independent accumulator chains interleave in the XDL pipe.
template<int KP>
__device__ inline void tile_mm3(const _Float16* __restrict__ Abase,
                                const _Float16* __restrict__ B0,
                                const _Float16* __restrict__ B1,
                                const _Float16* __restrict__ B2,
                                v8f& r0, v8f& r1, v8f& r2, int lane)
{
    const int g  = (lane >> 4) & 1;
    const int mr = lane & 15;
    const _Float16* ar = Abase + mr * KP;
    const _Float16* b0 = B0 + mr * KP;
    const _Float16* b1 = B1 + mr * KP;
    const _Float16* b2 = B2 + mr * KP;
#pragma unroll
    for (int q = 0; q < 8; ++q) { r0[q] = 0.f; r1[q] = 0.f; r2[q] = 0.f; }
#pragma unroll
    for (int k0 = 0; k0 < KP; k0 += 32) {
        v16h a, f0, f1, f2;
        frag_ld(ar, k0, g, a);
        frag_ld(b0, k0, g, f0);
        frag_ld(b1, k0, g, f1);
        frag_ld(b2, k0, g, f2);
        r0 = __builtin_amdgcn_wmma_f32_16x16x32_f16(false, a, false, f0, (short)0, r0, false, false);
        r1 = __builtin_amdgcn_wmma_f32_16x16x32_f16(false, a, false, f1, (short)0, r1, false, false);
        r2 = __builtin_amdgcn_wmma_f32_16x16x32_f16(false, a, false, f2, (short)0, r2, false, false);
    }
}

__device__ inline float sigf(float x) { return 1.f / (1.f + __expf(-x)); }

// ---------------------------------------------------------------------------
// Generic GEMM: out = act(A(MxKP,f16) @ W(NxKP,f16)^T + bias) [+ addend]
// outF/outH stride Nreal. act: 0 none, 1 tanh, 2 silu
// ---------------------------------------------------------------------------
template<int KP>
__global__ __launch_bounds__(256)
void gemm_wmma_kernel(const _Float16* __restrict__ A, const _Float16* __restrict__ W,
                      const float* __restrict__ bias, const float* __restrict__ addend,
                      float* __restrict__ outF, _Float16* __restrict__ outH,
                      int M, int N, int Nreal, int act)
{
    const int lane = threadIdx.x & 31, wave = threadIdx.x >> 5;
    const int tiles_n = N >> 4;
    const int tiles   = (M >> 4) * tiles_n;
    const int tile = blockIdx.x * 8 + wave;
    if (tile >= tiles) return;                 // wave-uniform exit (EXEC stays full)
    const int tm = tile / tiles_n, tn = tile % tiles_n;
    v8f acc = tile_mm<KP>(A + tm * 16 * KP, W + tn * 16 * KP, lane);
    const int g = lane >> 4, mr = lane & 15;
    const int n = tn * 16 + mr;
    if (n >= Nreal) return;
    const float bv = bias ? bias[n] : 0.f;
#pragma unroll
    for (int q = 0; q < 8; ++q) {
        const int m = tm * 16 + q + 8 * g;
        float v = acc[q] + bv;
        if (act == 1)      v = tanhf(v);
        else if (act == 2) v = v * sigf(v);
        if (addend) v += addend[m * Nreal + n];
        if (outF) outF[m * Nreal + n] = v;
        if (outH) outH[m * Nreal + n] = (_Float16)v;
    }
}

// ---------------------------------------------------------------------------
// Weight pack: f32 (Nr x Kr) -> f16 (Npad x Kpad), zero padded
// ---------------------------------------------------------------------------
__global__ void pack_w_kernel(const float* __restrict__ src, _Float16* __restrict__ dst,
                              int Nr, int Kr, int Np, int Kp)
{
    const int idx = blockIdx.x * 256 + threadIdx.x;
    if (idx >= Np * Kp) return;
    const int n = idx / Kp, k = idx % Kp;
    const float v = (n < Nr && k < Kr) ? src[n * Kr + k] : 0.f;
    dst[idx] = (_Float16)v;
}

// ---------------------------------------------------------------------------
// Fused STSGCN + time-MLP (+ node-MLP): one block per batch element.
// MODE 0: x path (src=x (30,256,63)), MODE 1: root (gather root_idx, V=1),
// MODE 2: cont (src=(256,60,NJ*4)).
// ---------------------------------------------------------------------------
template<int C, int T, int V, int MODE, bool NODE>
__global__ __launch_bounds__(256)
void sts_fused_kernel(const float* __restrict__ src, const int* __restrict__ ridx,
                      const float* __restrict__ Tm, const float* __restrict__ Am,
                      const float* __restrict__ convw, const float* __restrict__ convb,
                      const float* __restrict__ bng, const float* __restrict__ bnb,
                      const float* __restrict__ resw, const float* __restrict__ resb,
                      const float* __restrict__ rbng, const float* __restrict__ rbnb,
                      const float* __restrict__ prelu,
                      const float* __restrict__ tw1, const float* __restrict__ tb1,
                      const float* __restrict__ tw2, const float* __restrict__ tb2,
                      const float* __restrict__ nw1, const float* __restrict__ nb1,
                      const float* __restrict__ nw2, const float* __restrict__ nb2,
                      float* __restrict__ outF, _Float16* __restrict__ outH)
{
    constexpr int E = C * T * V;
    __shared__ float sx[E], sh[E], sh2[E];
    __shared__ float stw1[10 * T], stb1[10], stw2[10], stb2s[1];
    __shared__ float snw1[NODE ? 10 * V : 10], snb1[10], snw2[10], snb2s[1];
    const int n = blockIdx.x, tid = threadIdx.x;

    for (int i = tid; i < E; i += 256) {
        const int c = i / (T * V), r = i % (T * V), t = r / V, v = r % V;
        float val;
        if (MODE == 0)      val = src[(t * 256 + n) * 63 + v * 3 + c];
        else if (MODE == 1) val = src[(t * 256 + n) * 63 + ridx[c]];
        else                val = src[(n * 60 + t) * 84 + v * 4 + c];
        sx[i] = val;
    }
    for (int i = tid; i < 10 * T; i += 256) stw1[i] = tw1[i];
    if constexpr (NODE) { for (int i = tid; i < 10 * V; i += 256) snw1[i] = nw1[i]; }
    if (tid < 10) {
        stb1[tid] = tb1[tid]; stw2[tid] = tw2[tid];
        if constexpr (NODE) { snb1[tid] = nb1[tid]; snw2[tid] = nw2[tid]; }
    }
    if (tid == 0) { stb2s[0] = tb2[0]; if constexpr (NODE) snb2s[0] = nb2[0]; }
    __syncthreads();

    // h[c,q,v] = sum_t xi[c,t,v] * T[v,t,q]
    for (int i = tid; i < E; i += 256) {
        const int c = i / (T * V), r = i % (T * V), q = r / V, v = r % V;
        float acc = 0.f;
        for (int t = 0; t < T; ++t) acc += sx[c * T * V + t * V + v] * Tm[(v * T + t) * T + q];
        sh[i] = acc;
    }
    __syncthreads();
    // h2[c,t,w] = sum_v h[c,t,v] * A[t,v,w]
    for (int i = tid; i < E; i += 256) {
        const int c = i / (T * V), r = i % (T * V), t = r / V, w = r % V;
        float acc = 0.f;
        for (int v = 0; v < V; ++v) acc += sh[c * T * V + t * V + v] * Am[(t * V + v) * V + w];
        sh2[i] = acc;
    }
    __syncthreads();

    const int o = tid;
    float cw[C], rw[C];
#pragma unroll
    for (int c = 0; c < C; ++c) { cw[c] = convw[o * C + c]; rw[c] = resw[o * C + c]; }
    const float cb = convb[o], bg = bng[o] * BN_SCALE_C, bb = bnb[o];
    const float rb = resb[o], rg = rbng[o] * BN_SCALE_C, rbb = rbnb[o];
    const float pa = prelu[0];

    float tval[NODE ? V : 1];
    const int WN = NODE ? V : 1;
    for (int w = 0; w < WN; ++w) {
        float tacc[10];
#pragma unroll
        for (int j = 0; j < 10; ++j) tacc[j] = 0.f;
        for (int t = 0; t < T; ++t) {
            float s = 0.f, rr = 0.f;
#pragma unroll
            for (int c = 0; c < C; ++c) {
                s  += cw[c] * sh2[c * T * V + t * V + w];
                rr += rw[c] * sx [c * T * V + t * V + w];
            }
            float hv = bg * (s + cb) + bb + rg * (rr + rb) + rbb;
            hv = hv > 0.f ? hv : pa * hv;                     // PReLU
#pragma unroll
            for (int j = 0; j < 10; ++j) tacc[j] += stw1[j * T + t] * hv;
        }
        float hs = 0.f;
#pragma unroll
        for (int j = 0; j < 10; ++j) hs += tanhf(tacc[j] + stb1[j]) * stw2[j];
        tval[w] = tanhf(hs + stb2s[0]);
    }
    float outv;
    if constexpr (NODE) {
        float hs = 0.f;
#pragma unroll
        for (int j = 0; j < 10; ++j) {
            float a = snb1[j];
            for (int w = 0; w < V; ++w) a += snw1[j * V + w] * tval[w];
            hs += tanhf(a) * snw2[j];
        }
        outv = tanhf(hs + snb2s[0]);
    } else {
        outv = tval[0];
    }
    outF[n * 256 + o] = outv;
    if (outH) outH[n * 256 + o] = (_Float16)outv;
}

// ---------------------------------------------------------------------------
// root_his_dct[b, q*3+d] = sum_t dct[q,t] * x[min(t,29), b, root_idx[d]]
// ---------------------------------------------------------------------------
__global__ void root_dct_kernel(const float* __restrict__ x, const int* __restrict__ ridx,
                                const float* __restrict__ dct, float* __restrict__ rhd)
{
    const int b = blockIdx.x, i = threadIdx.x;
    if (i >= 180) return;
    const int q = i / 3, rj = ridx[i % 3];
    float acc = 0.f;
    for (int t = 0; t < 90; ++t) {
        const int ts = t < 30 ? t : 29;
        acc += dct[q * 90 + t] * x[(ts * 256 + b) * 63 + rj];
    }
    rhd[b * 180 + i] = acc;
}

// cat_h = f16 concat([root_his_dct(180), hx_root(256), hcont(256)], pad to 704)
__global__ void build_cat_kernel(const float* __restrict__ rhd, const float* __restrict__ hxr,
                                 const float* __restrict__ hc, _Float16* __restrict__ cat)
{
    const int idx = blockIdx.x * 256 + threadIdx.x;
    if (idx >= 256 * 704) return;
    const int b = idx / 704, c = idx % 704;
    float v = 0.f;
    if (c < 180)      v = rhd[b * 180 + c];
    else if (c < 436) v = hxr[b * 256 + (c - 180)];
    else if (c < 692) v = hc [b * 256 + (c - 436)];
    cat[idx] = (_Float16)v;
}

// pe_sel rows r=0..59 => pe[60-r]; 64x256, rows>=60 zero
__global__ void pe_kernel(_Float16* __restrict__ peh)
{
    const int idx = blockIdx.x * 256 + threadIdx.x; // 64*256
    const int r = idx >> 8, c = idx & 255;
    float v = 0.f;
    if (r < 60) {
        const float p   = (float)(60 - r);
        const int   j2  = c & ~1;
        const float div = __expf(-(float)j2 * 0.03597789207803196f); // ln(1e4)/256
        const float a   = p * div;
        v = (c & 1) ? __cosf(a) : __sinf(a);
    }
    peh[idx] = (_Float16)v;
}

// root_traj[b,t,d] = sum_q idct[t,q] * root_pred[b, q*3+d]
__global__ void idct_kernel(const float* __restrict__ idctm, const float* __restrict__ rp,
                            float* __restrict__ rt)
{
    const int b = blockIdx.x, i = threadIdx.x;
    if (i >= 270) return;
    const int t = i / 3, d = i % 3;
    float acc = 0.f;
    for (int q = 0; q < 60; ++q) acc += idctm[t * 60 + q] * rp[b * 180 + q * 3 + d];
    rt[b * 270 + i] = acc;
}

__global__ void scan_init_kernel(const float* __restrict__ x, float* __restrict__ ylast,
                                 unsigned* __restrict__ cnt)
{
    for (int i = threadIdx.x; i < 16128; i += 256) ylast[i] = x[29 * 16128 + i];
    if (threadIdx.x == 0) *cnt = 0u;
}

// ---------------------------------------------------------------------------
// Persistent GRU scan: 8 blocks x 256 threads, monotonic global barrier.
// ---------------------------------------------------------------------------
__device__ inline void gbar(unsigned* cnt, unsigned target)
{
    __syncthreads();
    if (threadIdx.x == 0) {
        __threadfence();
        atomicAdd(cnt, 1u);
        while (atomicAdd(cnt, 0u) < target) __builtin_amdgcn_s_sleep(2);
        __threadfence();
    }
    __syncthreads();
}

__global__ __launch_bounds__(256)
void gru_scan_kernel(const float* __restrict__ cont, const float* __restrict__ rt,
                     const float* __restrict__ te_f, const float* __restrict__ gh_f,
                     const float* __restrict__ hx_f,
                     const _Float16* __restrict__ Wo, const _Float16* __restrict__ Wih,
                     const _Float16* __restrict__ Wm,
                     const float* __restrict__ bo, const float* __restrict__ bih,
                     const float* __restrict__ bm,
                     _Float16* __restrict__ yext, _Float16* __restrict__ hy,
                     _Float16* __restrict__ hnew,
                     float* __restrict__ ylast, float* __restrict__ yout,
                     unsigned* __restrict__ cnt)
{
    const int NB = 8;
    const int tid = threadIdx.x, b = blockIdx.x;
    const int lane = tid & 31, wave = tid >> 5;
    const int wg = b * 8 + wave;                 // 0..63
    const int g = lane >> 4, mr = lane & 15;
    unsigned epoch = 0;

    for (int i = 0; i < 60; ++i) {
        // ---- phase 0: yext[b,256] = [ylast(63)|rstep(3)|end_root(3)|cstep(84)|end_cont(84)|0]
        for (int idx = b * 256 + tid; idx < 256 * 256; idx += 2048) {
            const int m = idx >> 8, c = idx & 255;
            float v;
            if (c < 63)       v = ylast[m * 63 + c];
            else if (c < 66)  v = rt[m * 270 + (30 + i) * 3 + (c - 63)];
            else if (c < 69)  v = rt[m * 270 + 89 * 3 + (c - 66)];
            else if (c < 153) v = cont[(m * 60 + i)  * 84 + (c - 69)];
            else if (c < 237) v = cont[(m * 60 + 59) * 84 + (c - 153)];
            else              v = 0.f;
            yext[idx] = (_Float16)v;
        }
        gbar(cnt, ++epoch * NB);

        // ---- phase 1: hy = yext @ Wo^T + bo + te[i]
        for (int t = wg; t < 256; t += 64) {
            const int tm = t >> 4, tn = t & 15;
            v8f acc = tile_mm<256>(yext + tm * 16 * 256, Wo + tn * 16 * 256, lane);
            const int n = tn * 16 + mr;
            const float add = bo[n] + te_f[i * 256 + n];
#pragma unroll
            for (int q = 0; q < 8; ++q) {
                const int m = tm * 16 + q + 8 * g;
                hy[m * 256 + n] = (_Float16)(acc[q] + add);
            }
        }
        gbar(cnt, ++epoch * NB);

        // ---- phase 2: gi = hy @ Wih^T + b_ih; gates; hnew  (block owns cols 32b..32b+31)
        for (int mm = 0; mm < 2; ++mm) {
            const int tm = wave * 2 + mm;
            for (int tc = 0; tc < 2; ++tc) {
                const int nb = b * 32 + tc * 16;
                v8f aR, aZ, aN;
                tile_mm3<256>(hy + tm * 16 * 256,
                              Wih + (      nb) * 256,
                              Wih + (256 + nb) * 256,
                              Wih + (512 + nb) * 256,
                              aR, aZ, aN, lane);
                const int n = nb + mr;
                const float b1 = bih[n], b2 = bih[256 + n], b3 = bih[512 + n];
#pragma unroll
                for (int q = 0; q < 8; ++q) {
                    const int m = tm * 16 + q + 8 * g;
                    const float r  = sigf(aR[q] + b1 + gh_f[m * 768 + n]);
                    const float z  = sigf(aZ[q] + b2 + gh_f[m * 768 + 256 + n]);
                    const float nn = tanhf(aN[q] + b3 + r * gh_f[m * 768 + 512 + n]);
                    const float h  = (1.f - z) * nn + z * hx_f[m * 256 + n];
                    hnew[m * 256 + n] = (_Float16)h;
                }
            }
        }
        gbar(cnt, ++epoch * NB);

        // ---- phase 3: ynew = ylast + hnew @ Wm^T + bm ; one tile per wave
        {
            const int tm = wg >> 2, tn = wg & 3;
            v8f acc = tile_mm<256>(hnew + tm * 16 * 256, Wm + tn * 16 * 256, lane);
            const int n = tn * 16 + mr;
            if (n < 63) {
                const float bb = bm[n];
#pragma unroll
                for (int q = 0; q < 8; ++q) {
                    const int m = tm * 16 + q + 8 * g;
                    const float v = acc[q] + bb + ylast[m * 63 + n];
                    ylast[m * 63 + n] = v;
                    yout[i * 16128 + m * 63 + n] = v;
                }
            }
        }
        gbar(cnt, ++epoch * NB);
    }
}

// ---------------------------------------------------------------------------
// Param indices: jax sorted-key flatten of setup_inputs() (84 inputs).
// ---------------------------------------------------------------------------
enum {
    IN_X = 0, IN_CONT = 1, IN_MASK = 2, IN_DCT = 3, IN_IDCT = 4, IN_RIDX = 5, IN_HOR = 6,
    P_CNE_B1 = 7, P_CNE_B2, P_CNE_W1, P_CNE_W2,
    P_CST_A = 11, P_CST_T, P_CST_BNB, P_CST_BNG, P_CST_CONVB, P_CST_CONVW,
    P_CST_PRELU, P_CST_RESB, P_CST_RESBNB, P_CST_RESBNG, P_CST_RESW,
    P_CTE_B1 = 22, P_CTE_B2, P_CTE_W1, P_CTE_W2,
    P_G1_B = 26, P_G1_W,
    P_G2_B1 = 28, P_G2_B2, P_G2_W1, P_G2_W2,
    P_G3_B1 = 32, P_G3_B2, P_G3_W1, P_G3_W2,
    P_G4_B = 36, P_G4_W,
    P_GRU_BHH = 38, P_GRU_BIH, P_GRU_WHH, P_GRU_WIH,
    P_OE_B = 42, P_OE_W,
    P_OM_B = 44, P_OM_W,
    P_RST_A = 46, P_RST_T, P_RST_BNB, P_RST_BNG, P_RST_CONVB, P_RST_CONVW,
    P_RST_PRELU, P_RST_RESB, P_RST_RESBNB, P_RST_RESBNG, P_RST_RESW,
    P_RTE_B1 = 57, P_RTE_B2, P_RTE_W1, P_RTE_W2,
    P_TE_B1 = 61, P_TE_B2, P_TE_W1, P_TE_W2,
    P_XNE_B1 = 65, P_XNE_B2, P_XNE_W1, P_XNE_W2,
    P_XST_A = 69, P_XST_T, P_XST_BNB, P_XST_BNG, P_XST_CONVB, P_XST_CONVW,
    P_XST_PRELU, P_XST_RESB, P_XST_RESBNB, P_XST_RESBNG, P_XST_RESW,
    P_XTE_B1 = 80, P_XTE_B2, P_XTE_W1, P_XTE_W2
};

extern "C" void kernel_launch(void* const* d_in, const int* in_sizes, int n_in,
                              void* d_out, int out_size, void* d_ws, size_t ws_size,
                              hipStream_t stream)
{
    (void)in_sizes; (void)n_in; (void)out_size; (void)ws_size;
    auto F = [&](int i) { return (const float*)d_in[i]; };
    const int* ridx = (const int*)d_in[IN_RIDX];

    // ---- workspace carve-up (256B aligned) ----
    char* base = (char*)d_ws;
    size_t off = 0;
    auto allocB = [&](size_t bytes) -> void* {
        void* p = base + off;
        off = (off + bytes + 255) & ~(size_t)255;
        return p;
    };
    auto H = [&](size_t n) { return (_Float16*)allocB(n * 2); };
    auto Fw = [&](size_t n) { return (float*)allocB(n * 4); };

    _Float16 *g1w_h = H(256 * 704);
    _Float16 *g2w1_h = H(65536), *g2w2_h = H(65536), *g3w1_h = H(65536), *g3w2_h = H(65536);
    _Float16 *g4w_h = H(192 * 256);
    _Float16 *tew1_h = H(65536), *tew2_h = H(65536);
    _Float16 *oew_h = H(65536), *wih_h = H(768 * 256), *whh_h = H(768 * 256), *wm_h = H(64 * 256);
    _Float16 *cat_h = H(256 * 704);
    _Float16 *h1_h = H(65536), *tmp_h = H(65536), *hx_h = H(65536);
    _Float16 *pe_h = H(64 * 256), *tetmp_h = H(64 * 256);
    _Float16 *yext_h = H(65536), *hy_h = H(65536), *hnew_h = H(65536);
    float *hx_f = Fw(65536), *hxroot_f = Fw(65536), *hcont_f = Fw(65536);
    float *rhd_f = Fw(256 * 180), *h1_f = Fw(65536), *rpred_f = Fw(256 * 180);
    float *gh_f = Fw(256 * 768), *te_f = Fw(64 * 256), *ylast_f = Fw(256 * 63);
    unsigned* cnt = (unsigned*)allocB(256);

    float* yout = (float*)d_out;                 // (60,256,63)
    float* rt   = (float*)d_out + 60 * 256 * 63; // (256,90,3)

    auto pack = [&](const float* s, _Float16* d, int Nr, int Kr, int Np, int Kp) {
        pack_w_kernel<<<dim3((Np * Kp + 255) / 256), dim3(256), 0, stream>>>(s, d, Nr, Kr, Np, Kp);
    };
    auto gemm = [&](const _Float16* A, const _Float16* W, const float* bias, const float* add,
                    float* oF, _Float16* oH, int M, int N, int Kp, int Nreal, int act) {
        const int tiles = (M / 16) * (N / 16);
        const dim3 grid((tiles + 7) / 8), blk(256);
        if (Kp == 704)
            gemm_wmma_kernel<704><<<grid, blk, 0, stream>>>(A, W, bias, add, oF, oH, M, N, Nreal, act);
        else
            gemm_wmma_kernel<256><<<grid, blk, 0, stream>>>(A, W, bias, add, oF, oH, M, N, Nreal, act);
    };

    // ---- weight packing (f32 -> f16, padded) ----
    pack(F(P_G1_W), g1w_h, 256, 692, 256, 704);
    pack(F(P_G2_W1), g2w1_h, 256, 256, 256, 256);
    pack(F(P_G2_W2), g2w2_h, 256, 256, 256, 256);
    pack(F(P_G3_W1), g3w1_h, 256, 256, 256, 256);
    pack(F(P_G3_W2), g3w2_h, 256, 256, 256, 256);
    pack(F(P_G4_W), g4w_h, 180, 256, 192, 256);
    pack(F(P_TE_W1), tew1_h, 256, 256, 256, 256);
    pack(F(P_TE_W2), tew2_h, 256, 256, 256, 256);
    pack(F(P_OE_W), oew_h, 256, 237, 256, 256);
    pack(F(P_GRU_WIH), wih_h, 768, 256, 768, 256);
    pack(F(P_GRU_WHH), whh_h, 768, 256, 768, 256);
    pack(F(P_OM_W), wm_h, 63, 256, 64, 256);

    // ---- fused STSGCN encoders ----
    sts_fused_kernel<3, 30, 21, 0, true><<<256, 256, 0, stream>>>(
        F(IN_X), nullptr, F(P_XST_T), F(P_XST_A),
        F(P_XST_CONVW), F(P_XST_CONVB), F(P_XST_BNG), F(P_XST_BNB),
        F(P_XST_RESW), F(P_XST_RESB), F(P_XST_RESBNG), F(P_XST_RESBNB), F(P_XST_PRELU),
        F(P_XTE_W1), F(P_XTE_B1), F(P_XTE_W2), F(P_XTE_B2),
        F(P_XNE_W1), F(P_XNE_B1), F(P_XNE_W2), F(P_XNE_B2), hx_f, hx_h);
    sts_fused_kernel<3, 30, 1, 1, false><<<256, 256, 0, stream>>>(
        F(IN_X), ridx, F(P_RST_T), F(P_RST_A),
        F(P_RST_CONVW), F(P_RST_CONVB), F(P_RST_BNG), F(P_RST_BNB),
        F(P_RST_RESW), F(P_RST_RESB), F(P_RST_RESBNG), F(P_RST_RESBNB), F(P_RST_PRELU),
        F(P_RTE_W1), F(P_RTE_B1), F(P_RTE_W2), F(P_RTE_B2),
        nullptr, nullptr, nullptr, nullptr, hxroot_f, nullptr);
    sts_fused_kernel<4, 60, 21, 2, true><<<256, 256, 0, stream>>>(
        F(IN_CONT), nullptr, F(P_CST_T), F(P_CST_A),
        F(P_CST_CONVW), F(P_CST_CONVB), F(P_CST_BNG), F(P_CST_BNB),
        F(P_CST_RESW), F(P_CST_RESB), F(P_CST_RESBNG), F(P_CST_RESBNB), F(P_CST_PRELU),
        F(P_CTE_W1), F(P_CTE_B1), F(P_CTE_W2), F(P_CTE_B2),
        F(P_CNE_W1), F(P_CNE_B1), F(P_CNE_W2), F(P_CNE_B2), hcont_f, nullptr);

    // ---- DCT of root history, concat, time embedding ----
    root_dct_kernel<<<256, 192, 0, stream>>>(F(IN_X), ridx, F(IN_DCT), rhd_f);
    build_cat_kernel<<<704, 256, 0, stream>>>(rhd_f, hxroot_f, hcont_f, cat_h);
    pe_kernel<<<64, 256, 0, stream>>>(pe_h);
    gemm(pe_h, tew1_h, F(P_TE_B1), nullptr, nullptr, tetmp_h, 64, 256, 256, 256, 2); // silu
    gemm(tetmp_h, tew2_h, F(P_TE_B2), nullptr, te_f, nullptr, 64, 256, 256, 256, 0);

    // ---- g-MLP chain (WMMA) ----
    gemm(cat_h, g1w_h, F(P_G1_B), nullptr, h1_f, h1_h, 256, 256, 704, 256, 1);        // tanh
    gemm(h1_h, g2w1_h, F(P_G2_B1), nullptr, nullptr, tmp_h, 256, 256, 256, 256, 1);
    gemm(tmp_h, g2w2_h, F(P_G2_B2), h1_f, h1_f, h1_h, 256, 256, 256, 256, 1);         // +res
    gemm(h1_h, g3w1_h, F(P_G3_B1), nullptr, nullptr, tmp_h, 256, 256, 256, 256, 1);
    gemm(tmp_h, g3w2_h, F(P_G3_B2), h1_f, h1_f, h1_h, 256, 256, 256, 256, 1);         // +res
    gemm(h1_h, g4w_h, F(P_G4_B), rhd_f, rpred_f, nullptr, 256, 192, 256, 180, 0);     // +dct

    // ---- gh = hx @ w_hh^T + b_hh ----
    gemm(hx_h, whh_h, F(P_GRU_BHH), nullptr, gh_f, nullptr, 256, 768, 256, 768, 0);

    // ---- iDCT -> root_traj (output 2) ----
    idct_kernel<<<256, 288, 0, stream>>>(F(IN_IDCT), rpred_f, rt);

    // ---- persistent GRU scan (output 1) ----
    scan_init_kernel<<<1, 256, 0, stream>>>(F(IN_X), ylast_f, cnt);
    gru_scan_kernel<<<8, 256, 0, stream>>>(
        F(IN_CONT), rt, te_f, gh_f, hx_f,
        oew_h, wih_h, wm_h,
        F(P_OE_B), F(P_GRU_BIH), F(P_OM_B),
        yext_h, hy_h, hnew_h, ylast_f, yout, cnt);
}
